// MhAttnBlock_21225728377477
// MI455X (gfx1250) — matrix operand order repaired and checked by reading, not compiled, not run on hardware
//
#include <hip/hip_runtime.h>
#include <hip/hip_bf16.h>
#include <math.h>

// ---- problem constants (match reference) ----
#define B_    8
#define QS_   1024
#define KS_   1024
#define DIN   256          // QDIM = KDIM = VDIM
#define H_    8
#define OD    64
#define NPROJ (H_ * OD)    // 512
#define NEG_SLOPE 0.2f

typedef __bf16 bf16_t;
typedef bf16_t v16bf __attribute__((ext_vector_type(16)));
typedef bf16_t v8bf  __attribute__((ext_vector_type(8)));
typedef float  v8f   __attribute__((ext_vector_type(8)));
typedef float  v4f   __attribute__((ext_vector_type(4)));

static __device__ __forceinline__ v16bf join8(v8bf lo, v8bf hi) {
    return __builtin_shufflevector(lo, hi, 0,1,2,3,4,5,6,7,8,9,10,11,12,13,14,15);
}

// ============================================================================
// Stage 0: WT[which][n][k] = bf16(W[k][n])  -- transpose+convert weights once
// so stage-1 B fragments are contiguous b128 loads.
// ============================================================================
__global__ __launch_bounds__(256)
void wcvt_kernel(const float* __restrict__ WQ, const float* __restrict__ WK,
                 const float* __restrict__ WV, bf16_t* __restrict__ WT)
{
    int idx = blockIdx.x * 256 + threadIdx.x;        // 3*512*256
    if (idx >= 3 * NPROJ * DIN) return;
    const int which = idx / (NPROJ * DIN);
    const int r     = idx % (NPROJ * DIN);
    const int n     = r / DIN;
    const int k     = r % DIN;
    const float* W  = (which == 0) ? WQ : (which == 1) ? WK : WV;
    WT[idx] = (bf16_t)W[(size_t)k * NPROJ + n];
}

// ============================================================================
// Stage 1: projection GEMM  Xp = X @ W   (X:[8192,256] f32, WT bf16 [n][k])
// One wave computes a 16x64 tile.  A frag from contiguous f32 (v4f loads),
// B frag = two b128 bf16 loads per 16x16 n-tile.
// B layout (ISA 7.12.5 dense analog): lane<16: N=lane,  K={kc+0..7, kc+16..23}
//                                     lane>=16: N=lane-16, K={kc+8..15, kc+24..31}
// A layout (ISA 7.12.2):              lane<16: M=lane,  K={kc+0..7, kc+16..23}
//                                     lane>=16: M=lane-16, K={kc+8..15, kc+24..31}
// Q/K projections stored f32 row-major; V projection stored bf16 TRANSPOSED:
// VpT[b][d][k]  (so stage-3 B fragments are contiguous b128 loads too).
// ============================================================================
__global__ __launch_bounds__(32)
void proj_gemm_kernel(const float* __restrict__ Q, const float* __restrict__ K,
                      const float* __restrict__ V,
                      const bf16_t* __restrict__ WT,
                      float* __restrict__ Qp, float* __restrict__ Kp,
                      bf16_t* __restrict__ VpT)
{
    const int lane = threadIdx.x & 31;
    const int hf   = lane >> 4;
    const int l15  = lane & 15;

    const int mbase = blockIdx.x * 16;   // over M = 8192
    const int nbase = blockIdx.y * 64;   // over N = 512
    const int which = blockIdx.z;

    const float* X = (which == 0) ? Q : (which == 1) ? K : V;

    v8f acc[4] = {};

    const int arow = mbase + l15;
    const float* aprow = X + (size_t)arow * DIN;

    for (int kc = 0; kc < DIN; kc += 32) {
        // ---- A fragment: two contiguous 8-float runs, cvt to bf16 ----
        const float* ap = aprow + kc + hf * 8;
        v4f x0 = *(const v4f*)(ap);
        v4f x1 = *(const v4f*)(ap + 4);
        v4f x2 = *(const v4f*)(ap + 16);
        v4f x3 = *(const v4f*)(ap + 20);
        v16bf a;
#pragma unroll
        for (int e = 0; e < 4; ++e) {
            a[e]      = (bf16_t)x0[e];
            a[e + 4]  = (bf16_t)x1[e];
            a[e + 8]  = (bf16_t)x2[e];
            a[e + 12] = (bf16_t)x3[e];
        }
        // ---- 4 B fragments + WMMA ----
#pragma unroll
        for (int t = 0; t < 4; ++t) {
            const int ncol = nbase + t * 16 + l15;
            const bf16_t* bp =
                WT + ((size_t)which * NPROJ + ncol) * DIN + kc + hf * 8;
            v16bf bfrag = join8(*(const v8bf*)bp, *(const v8bf*)(bp + 16));
            acc[t] = __builtin_amdgcn_wmma_f32_16x16x32_bf16(
                false, a, false, bfrag, (short)0, acc[t], false, false);
        }
    }

    // ---- store: Q/K -> f32 [m][n]; V -> bf16 transposed [b][d][k] ----
#pragma unroll
    for (int t = 0; t < 4; ++t) {
        const int ncol = nbase + t * 16 + l15;
#pragma unroll
        for (int r = 0; r < 8; ++r) {
            const int m = mbase + r + hf * 8;        // C/D row
            if (which == 0) {
                Qp[(size_t)m * NPROJ + ncol] = acc[t][r];
            } else if (which == 1) {
                Kp[(size_t)m * NPROJ + ncol] = acc[t][r];
            } else {
                const int b = m >> 10;               // m / KS_
                const int k = m & (KS_ - 1);
                VpT[((size_t)b * NPROJ + ncol) * KS_ + k] = (bf16_t)acc[t][r];
            }
        }
    }
}

// ============================================================================
// Stage 2: attn logits, stored TRANSPOSED [b][h][s] for contiguous stage-3 reads
// ============================================================================
__global__ __launch_bounds__(256)
void attn_sum_kernel(const float* __restrict__ Qp, const float* __restrict__ Kp,
                     const float* __restrict__ alphaQ, const float* __restrict__ alphaK,
                     float* __restrict__ attn_q_t, float* __restrict__ attn_k_t)
{
    const int total = B_ * QS_ * H_;
    int idx = blockIdx.x * blockDim.x + threadIdx.x;
    if (idx >= 2 * total) return;
    const int sel = (idx >= total);
    const int r   = idx - sel * total;
    const int h   = r % H_;
    const int s   = (r / H_) % QS_;
    const int b   = r / (H_ * QS_);

    const float* Xp  = sel ? Kp : Qp;
    const float* al  = (sel ? alphaK : alphaQ) + h * OD;
    const float* row = Xp + ((size_t)(b * QS_ + s)) * NPROJ + h * OD;

    float sum = 0.f;
#pragma unroll 8
    for (int d = 0; d < OD; ++d) sum = fmaf(al[d], row[d], sum);
    (sel ? attn_k_t : attn_q_t)[((size_t)(b * H_ + h)) * QS_ + s] = sum;
}

// ============================================================================
// Stage 2b: akmax[b,h] = max_k attn_k_t[b,h,k]   (contiguous reduce)
// ============================================================================
__global__ __launch_bounds__(256)
void akmax_kernel(const float* __restrict__ attn_k_t, float* __restrict__ akmax)
{
    __shared__ float red[256];
    const int bh = blockIdx.x;
    const float* base = attn_k_t + (size_t)bh * KS_;
    float m = -3.4e38f;
    for (int k = threadIdx.x; k < KS_; k += 256) m = fmaxf(m, base[k]);
    red[threadIdx.x] = m;
    __syncthreads();
    for (int s = 128; s > 0; s >>= 1) {
        if (threadIdx.x < s)
            red[threadIdx.x] = fmaxf(red[threadIdx.x], red[threadIdx.x + s]);
        __syncthreads();
    }
    if (threadIdx.x == 0) akmax[bh] = red[0];
}

// ============================================================================
// Stage 3: fused softmax + P@V.  One wave handles TWO 16-query tiles (shares
// attn_k and B-fragment loads), producing 32x64 output per wave.
//   max_k lrelu(aq+ak[k]) = lrelu(aq + max_k ak[k])  (lrelu monotone) -> exact
//   shift with no score-matrix pass.  Masked entries -> p = 0.
// ============================================================================
#define QT_ 2
__global__ __launch_bounds__(32)
void attn_wmma_kernel(const bf16_t* __restrict__ VpT,
                      const float* __restrict__ attn_q_t,
                      const float* __restrict__ attn_k_t,
                      const float* __restrict__ akmax,
                      const unsigned char* __restrict__ mask,
                      const float* __restrict__ bias,
                      float* __restrict__ out)
{
    const int lane = threadIdx.x & 31;
    const int hf   = lane >> 4;
    const int l15  = lane & 15;

    const int qbase = blockIdx.x * (16 * QT_);   // 32 tiles of 32 queries
    const int h     = blockIdx.y;
    const int b     = blockIdx.z;

    const float* akb = attn_k_t + ((size_t)(b * H_ + h)) * KS_;
    const float akm  = akmax[b * H_ + h];

    float aq[QT_], mrow[QT_], dsum[QT_];
    const unsigned char* mrp[QT_];
#pragma unroll
    for (int qt = 0; qt < QT_; ++qt) {
        const int qrow = qbase + qt * 16 + l15;
        aq[qt] = attn_q_t[((size_t)(b * H_ + h)) * QS_ + qrow];
        const float mi = aq[qt] + akm;
        mrow[qt] = (mi > 0.f) ? mi : NEG_SLOPE * mi;
        dsum[qt] = 0.f;
        mrp[qt]  = mask + ((size_t)b * QS_ + qrow) * KS_;
    }

    v8f acc[QT_][4] = {};

    for (int kc = 0; kc < KS_; kc += 32) {
        const int k0 = kc + hf * 8;              // A-frag K pattern base
        // shared logit loads (contiguous b128)
        v4f ak0 = *(const v4f*)(akb + k0);
        v4f ak1 = *(const v4f*)(akb + k0 + 4);
        v4f ak2 = *(const v4f*)(akb + k0 + 16);
        v4f ak3 = *(const v4f*)(akb + k0 + 20);

        v16bf afr[QT_];
#pragma unroll
        for (int qt = 0; qt < QT_; ++qt) {
            const unsigned long long m0 = *(const unsigned long long*)(mrp[qt] + k0);
            const unsigned long long m1 = *(const unsigned long long*)(mrp[qt] + k0 + 16);
#pragma unroll
            for (int e = 0; e < 4; ++e) {
                float s, lr, p;
                s  = aq[qt] + ak0[e];
                lr = (s > 0.f) ? s : NEG_SLOPE * s;
                p  = ((m0 >> (8 * e)) & 0xffull) ? __expf(lr - mrow[qt]) : 0.f;
                dsum[qt] += p;  afr[qt][e] = (bf16_t)p;

                s  = aq[qt] + ak1[e];
                lr = (s > 0.f) ? s : NEG_SLOPE * s;
                p  = ((m0 >> (8 * (e + 4))) & 0xffull) ? __expf(lr - mrow[qt]) : 0.f;
                dsum[qt] += p;  afr[qt][e + 4] = (bf16_t)p;

                s  = aq[qt] + ak2[e];
                lr = (s > 0.f) ? s : NEG_SLOPE * s;
                p  = ((m1 >> (8 * e)) & 0xffull) ? __expf(lr - mrow[qt]) : 0.f;
                dsum[qt] += p;  afr[qt][e + 8] = (bf16_t)p;

                s  = aq[qt] + ak3[e];
                lr = (s > 0.f) ? s : NEG_SLOPE * s;
                p  = ((m1 >> (8 * (e + 4))) & 0xffull) ? __expf(lr - mrow[qt]) : 0.f;
                dsum[qt] += p;  afr[qt][e + 12] = (bf16_t)p;
            }
        }
        // B fragments (shared across both q-tiles): contiguous bf16 in VpT
#pragma unroll
        for (int t = 0; t < 4; ++t) {
            const int d = h * OD + t * 16 + l15;
            const bf16_t* bp =
                VpT + ((size_t)b * NPROJ + d) * KS_ + kc + hf * 8;
            v16bf bfrag = join8(*(const v8bf*)bp, *(const v8bf*)(bp + 16));
#pragma unroll
            for (int qt = 0; qt < QT_; ++qt)
                acc[qt][t] = __builtin_amdgcn_wmma_f32_16x16x32_bf16(
                    false, afr[qt], false, bfrag, (short)0, acc[qt][t], false, false);
        }
    }

    // merge split-row denominators (lane L and L^16 cover complementary K)
    __shared__ float sden[QT_][16];
#pragma unroll
    for (int qt = 0; qt < QT_; ++qt) {
        dsum[qt] += __shfl_xor(dsum[qt], 16, 32);
        if (lane < 16) sden[qt][lane] = dsum[qt];
    }
    __syncthreads();

#pragma unroll
    for (int qt = 0; qt < QT_; ++qt) {
#pragma unroll
        for (int t = 0; t < 4; ++t) {
            const int dcol = t * 16 + l15;
            const float bv = bias[h * OD + dcol];
#pragma unroll
            for (int r = 0; r < 8; ++r) {
                const int m = r + hf * 8;
                const float inv = 1.0f / sden[qt][m];
                out[((size_t)(b * QS_ + qbase + qt * 16 + m)) * NPROJ + h * OD + dcol] =
                    acc[qt][t][r] * inv + bv;
            }
        }
    }
}

// ============================================================================
extern "C" void kernel_launch(void* const* d_in, const int* in_sizes, int n_in,
                              void* d_out, int out_size, void* d_ws, size_t ws_size,
                              hipStream_t stream)
{
    const float*         Q    = (const float*)d_in[0];
    const float*         K    = (const float*)d_in[1];
    const float*         V    = (const float*)d_in[2];
    const unsigned char* mask = (const unsigned char*)d_in[3];
    const float*         WQ   = (const float*)d_in[4];
    const float*         WK   = (const float*)d_in[5];
    const float*         WV   = (const float*)d_in[6];
    const float*         aQ   = (const float*)d_in[7];
    const float*         aK   = (const float*)d_in[8];
    const float*         bias = (const float*)d_in[9];
    float*               out  = (float*)d_out;

    // workspace: [Qp f32 | Kp f32 | attn_q | attn_k | akmax | VpT bf16 | WT bf16]
    float* ws = (float*)d_ws;
    const size_t projElems = (size_t)B_ * QS_ * NPROJ;       // 4,194,304
    float*  Qp     = ws;
    float*  Kp     = Qp + projElems;
    float*  attn_q = Kp + projElems;
    float*  attn_k = attn_q + (size_t)B_ * QS_ * H_;
    float*  akmax  = attn_k + (size_t)B_ * KS_ * H_;
    bf16_t* VpT    = (bf16_t*)(akmax + 64);                  // 16B-aligned
    bf16_t* WT     = VpT + projElems;                        // [3][512][256]

    // Stage 0: weight transpose+convert
    const int wtot = 3 * NPROJ * DIN;
    wcvt_kernel<<<(wtot + 255) / 256, 256, 0, stream>>>(WQ, WK, WV, WT);

    // Stage 1: three projection GEMMs
    dim3 g1((B_ * QS_) / 16, NPROJ / 64, 3);
    proj_gemm_kernel<<<g1, 32, 0, stream>>>(Q, K, V, WT, Qp, Kp, VpT);

    // Stage 2: attention logits (transposed layout)
    const int total2 = 2 * B_ * QS_ * H_;
    attn_sum_kernel<<<(total2 + 255) / 256, 256, 0, stream>>>(
        Qp, Kp, aQ, aK, attn_q, attn_k);

    // Stage 2b: key-side max per (b,h)
    akmax_kernel<<<B_ * H_, 256, 0, stream>>>(attn_k, akmax);

    // Stage 3: fused softmax + P@V
    dim3 g3(QS_ / (16 * QT_), H_, B_);                       // 32 x 8 x 8
    attn_wmma_kernel<<<g3, 32, 0, stream>>>(
        VpT, attn_q, attn_k, akmax, mask, bias, out);
}
